// HungarianMatcher_21423296873282
// MI455X (gfx1250) — matrix operand order repaired and checked by reading, not compiled, not run on hardware
//
#include <hip/hip_runtime.h>
#include <hip/hip_bf16.h>

// ---------------- problem constants (from reference setup_inputs) ----------
#define BB 4
#define QQ 200
#define QT 13          // ceil(200/16)
#define QP 208         // padded Q rows in workspace
#define NN 32
#define CC 80
#define HWD 65536      // 256*256
#define KSLICES 64     // K split per (b, qtile); 1024 K per wave
#define KLEN (HWD / KSLICES)

#define LOG2E 1.44269504088896340736f
#define LN2   0.69314718055994530942f

// ---------------- workspace layout (floats) --------------------------------
#define OFF_DOT_S   0
#define OFF_DOT_X   (BB*QP*NN)
#define OFF_SUM_S   (2*BB*QP*NN)
#define OFF_SUM_LNS (2*BB*QP*NN + BB*QP)
#define OFF_SUM_T   (2*BB*QP*NN + 2*BB*QP)
#define WS_FLOATS   (2*BB*QP*NN + 2*BB*QP + BB*NN)

typedef __attribute__((ext_vector_type(16))) __bf16 v16bf;
typedef __attribute__((ext_vector_type(8)))  float  v8f;

// ============================================================================
// Main fused kernel: one wave32 owns (b, 16-row Q tile, K slice) and BOTH
// 16-col N tiles.  Per 32-wide K step:
//   A fragment (16x32): lane half 0 -> K {0..15},
//   lane half 1 -> K {16..31}; row M = lane&15.  Four aligned float4s.
//   Transform: e = exp2(-x*log2e); s = rcp(1+e); clamp via med3;
//              log_s = log2(s)*ln2; log_ns = log_s - x;
//              bce A-matrix is bf16(x) directly (log_s - log_ns == x).
//   B fragment (32x16): lane half selects K range [16h,16h+16), col N = lane&15
//   -> 16 contiguous floats per lane per N tile.
//   4x v_wmma_f32_16x16x32_bf16 accumulate in f32.
// Partials land in workspace via global_atomic_add_f32.
// ============================================================================
__global__ __launch_bounds__(256) void cost_gemm_kernel(
    const float* __restrict__ pred_masks,  // [B,Q,HW]
    const float* __restrict__ gt_masks,    // [B,N,HW]
    float* __restrict__ ws)
{
  const int wave = (int)((blockIdx.x * blockDim.x + threadIdx.x) >> 5);
  const int lane = (int)(threadIdx.x & 31);
  const int ks   = wave % KSLICES;
  const int t2   = wave / KSLICES;
  const int qt   = t2 % QT;
  const int b    = t2 / QT;
  const int half = lane >> 4;
  const int r    = lane & 15;

  // row used for loads (clamped for the partial last Q tile; results for
  // padded rows land in padded workspace rows and are never read)
  int qload = qt * 16 + r;
  if (qload > QQ - 1) qload = QQ - 1;

  const float* __restrict__ pA  = pred_masks + (size_t)(b * QQ + qload) * HWD;
  const float* __restrict__ pB0 = gt_masks   + (size_t)(b * NN + r)      * HWD;
  const float* __restrict__ pB1 = gt_masks   + (size_t)(b * NN + 16 + r) * HWD;

  v8f accS0 = {}; v8f accS1 = {};   // dot(s, t)  for N tiles 0,1
  v8f accX0 = {}; v8f accX1 = {};   // dot(x, t)  for N tiles 0,1
  float rs_s = 0.f, rs_lns = 0.f;   // per-row partial sums (this lane's K)

  const int abase = half * 16;      // A-fragment K base within 32-chunk
  const int bbase = half * 16;      // B-fragment K base within 32-chunk
  const int k_begin = ks * KLEN;
  const int k_end   = k_begin + KLEN;

  for (int k0 = k_begin; k0 < k_end; k0 += 32) {
    // ---- A: 16 pred-mask logits this lane owns --------------------------
    float4 a0 = *(const float4*)(pA + k0 + abase);
    float4 a1 = *(const float4*)(pA + k0 + abase + 4);
    float4 a2 = *(const float4*)(pA + k0 + abase + 8);
    float4 a3 = *(const float4*)(pA + k0 + abase + 12);
    float xv[16] = {a0.x,a0.y,a0.z,a0.w, a1.x,a1.y,a1.z,a1.w,
                    a2.x,a2.y,a2.z,a2.w, a3.x,a3.y,a3.z,a3.w};
    v16bf aS, aX;
#pragma unroll
    for (int j = 0; j < 16; ++j) {
      const float x = xv[j];
      // raw v_exp_f32 / v_rcp_f32: operands never denormal/inf here
      const float e = __builtin_amdgcn_exp2f(x * -LOG2E);
      float s = __builtin_amdgcn_rcpf(1.0f + e);
      s = fminf(fmaxf(s, 1e-6f), 1.0f - 1e-6f);           // single v_med3
      // raw v_log_f32 (log2); s >= 1e-6 so no denorm fixup needed
      const float ls  = __builtin_amdgcn_logf(s) * LN2;   // log(s)
      const float lns = ls - x;                           // log(1-s)
      rs_s   += s;
      rs_lns += lns;
      aS[j] = (__bf16)s;
      aX[j] = (__bf16)x;                                  // log_s - log_ns == x
    }

    // ---- B: gt masks, 16 contiguous values per N tile -------------------
    float4 u0 = *(const float4*)(pB0 + k0 + bbase);
    float4 u1 = *(const float4*)(pB0 + k0 + bbase + 4);
    float4 u2 = *(const float4*)(pB0 + k0 + bbase + 8);
    float4 u3 = *(const float4*)(pB0 + k0 + bbase + 12);
    float4 w0 = *(const float4*)(pB1 + k0 + bbase);
    float4 w1 = *(const float4*)(pB1 + k0 + bbase + 4);
    float4 w2 = *(const float4*)(pB1 + k0 + bbase + 8);
    float4 w3 = *(const float4*)(pB1 + k0 + bbase + 12);
    float tv0[16] = {u0.x,u0.y,u0.z,u0.w, u1.x,u1.y,u1.z,u1.w,
                     u2.x,u2.y,u2.z,u2.w, u3.x,u3.y,u3.z,u3.w};
    float tv1[16] = {w0.x,w0.y,w0.z,w0.w, w1.x,w1.y,w1.z,w1.w,
                     w2.x,w2.y,w2.z,w2.w, w3.x,w3.y,w3.z,w3.w};
    v16bf bT0, bT1;
#pragma unroll
    for (int j = 0; j < 16; ++j) {
      bT0[j] = (__bf16)tv0[j];
      bT1[j] = (__bf16)tv1[j];
    }

    // ---- 4 WMMAs: 2 A-matrices x 2 N tiles ------------------------------
    accS0 = __builtin_amdgcn_wmma_f32_16x16x32_bf16(false, aS, false, bT0,
                                                    (short)0, accS0, false, false);
    accX0 = __builtin_amdgcn_wmma_f32_16x16x32_bf16(false, aX, false, bT0,
                                                    (short)0, accX0, false, false);
    accS1 = __builtin_amdgcn_wmma_f32_16x16x32_bf16(false, aS, false, bT1,
                                                    (short)0, accS1, false, false);
    accX1 = __builtin_amdgcn_wmma_f32_16x16x32_bf16(false, aX, false, bT1,
                                                    (short)0, accX1, false, false);
  }

  // ---- combine partials into workspace ----------------------------------
  // f32 C/D layout: VGPR v, lanes 0-15 -> (M=v, N=lane); lanes 16-31 -> (M=8+v)
  float* dotS = ws + OFF_DOT_S + (size_t)b * QP * NN;
  float* dotX = ws + OFF_DOT_X + (size_t)b * QP * NN;
#pragma unroll
  for (int v = 0; v < 8; ++v) {
    const int row = qt * 16 + half * 8 + v;   // < QP always
    atomicAdd(&dotS[row * NN + r],      accS0[v]);
    atomicAdd(&dotS[row * NN + 16 + r], accS1[v]);
    atomicAdd(&dotX[row * NN + r],      accX0[v]);
    atomicAdd(&dotX[row * NN + 16 + r], accX1[v]);
  }
  const int rrow = qt * 16 + r;               // lanes r and r+16 both add
  atomicAdd(&ws[OFF_SUM_S   + b * QP + rrow], rs_s);
  atomicAdd(&ws[OFF_SUM_LNS + b * QP + rrow], rs_lns);
}

// ============================================================================
// sum_t: reduce gt_masks over HW -> [B*N]
// ============================================================================
__global__ __launch_bounds__(256) void sum_t_kernel(
    const float* __restrict__ gt_masks, float* __restrict__ sum_t)
{
  __shared__ float red[256];
  const int bn = blockIdx.x;                       // 0 .. B*N-1
  const float4* p = (const float4*)(gt_masks + (size_t)bn * HWD);
  float acc = 0.f;
  for (int i = threadIdx.x; i < HWD / 4; i += 256) {
    float4 v = p[i];
    acc += v.x + v.y + v.z + v.w;
  }
  red[threadIdx.x] = acc;
  __syncthreads();
  for (int s = 128; s > 0; s >>= 1) {
    if ((int)threadIdx.x < s) red[threadIdx.x] += red[threadIdx.x + s];
    __syncthreads();
  }
  if (threadIdx.x == 0) sum_t[bn] = red[0];
}

// ============================================================================
// finalize: class-cost gather + bce/dice assembly + finite fill -> [B,Q,N]
// ============================================================================
__global__ __launch_bounds__(256) void finalize_kernel(
    const float* __restrict__ pred_logits,   // [B,Q,C]
    const int*   __restrict__ gt_classes,    // [B,N]
    const float* __restrict__ ws,
    float* __restrict__ out)                 // [B,Q,N]
{
  const int idx = blockIdx.x * blockDim.x + threadIdx.x;
  if (idx >= BB * QQ * NN) return;
  const int n = idx % NN;
  const int q = (idx / NN) % QQ;
  const int b = idx / (NN * QQ);

  const int   cls   = gt_classes[b * NN + n];
  const float logit = pred_logits[(b * QQ + q) * CC + cls];
  const float prob  =
      __builtin_amdgcn_rcpf(1.0f + __builtin_amdgcn_exp2f(logit * -LOG2E));
  const float cost_class = -prob;

  const float dx   = ws[OFF_DOT_X + ((size_t)b * QP + q) * NN + n];
  const float lns  = ws[OFF_SUM_LNS + b * QP + q];
  const float cost_bce = -(dx + lns) * (1.0f / (float)HWD);

  const float ds = ws[OFF_DOT_S + ((size_t)b * QP + q) * NN + n];
  const float ss = ws[OFF_SUM_S + b * QP + q];
  const float st = ws[OFF_SUM_T + b * NN + n];
  const float cost_dice = 1.0f - 2.0f * ds / (ss + st + 1e-6f);

  float Cv = 1.0f * cost_class + 5.0f * cost_bce + 5.0f * cost_dice;
  if (!(Cv == Cv) || fabsf(Cv) > 3.0e38f) Cv = 10000.0f;  // isfinite fill
  out[idx] = Cv;
}

// ============================================================================
extern "C" void kernel_launch(void* const* d_in, const int* in_sizes, int n_in,
                              void* d_out, int out_size, void* d_ws, size_t ws_size,
                              hipStream_t stream) {
  const float* pred_logits = (const float*)d_in[0];   // [4,200,80]      f32
  const float* pred_masks  = (const float*)d_in[1];   // [4,200,256,256] f32
  const int*   gt_classes  = (const int*)  d_in[2];   // [4,32]          i32
  const float* gt_masks    = (const float*)d_in[3];   // [4,32,256,256]  f32
  float* ws  = (float*)d_ws;
  float* out = (float*)d_out;

  // workspace is accumulated into atomically -> must be zeroed every call
  hipMemsetAsync(d_ws, 0, (size_t)WS_FLOATS * sizeof(float), stream);

  // column sums of t
  sum_t_kernel<<<BB * NN, 256, 0, stream>>>(gt_masks, ws + OFF_SUM_T);

  // fused transform + dual bf16 WMMA GEMM + row sums
  const int total_waves = BB * QT * KSLICES;          // 3328
  const int blocks = total_waves / 8;                 // 256 thr = 8 wave32
  cost_gemm_kernel<<<blocks, 256, 0, stream>>>(pred_masks, gt_masks, ws);

  // assemble cost matrix
  const int total = BB * QQ * NN;
  finalize_kernel<<<(total + 255) / 256, 256, 0, stream>>>(
      pred_logits, gt_classes, ws, out);
}
